// Input_embedding_35811437314388
// MI455X (gfx1250) — matrix operand order repaired and checked by reading, not compiled, not run on hardware
//
#include <hip/hip_runtime.h>
#include <hip/hip_bf16.h>

// ---------------- problem constants (from reference) ----------------
#define BB   4
#define NN   8192
#define CO   64
#define KNB  20            // neighbors kept
#define RTOT (BB * NN * KNB)   // 655360 rows of edge features
#define RT16 (RTOT / 16)       // 40960 row tiles (exact)
#define EPS  1e-5f
#define SLOPE 0.1f

typedef __attribute__((ext_vector_type(16))) _Float16 v16h;
typedef __attribute__((ext_vector_type(8)))  float    v8f;
typedef unsigned int u32x4 __attribute__((ext_vector_type(4)));
typedef int          i32x4 __attribute__((ext_vector_type(4)));
typedef int          i32x8 __attribute__((ext_vector_type(8)));

__device__ __forceinline__ v8f wmma_f16(v16h a, v16h b, v8f c) {
  // D = A(16x32 f16) * B(32x16 f16) + C(16x16 f32)
  return __builtin_amdgcn_wmma_f32_16x16x32_f16(
      /*neg_a=*/false, a, /*neg_b=*/false, b,
      /*c_mod=*/(short)0, c, /*reuse_a=*/false, /*reuse_b=*/false);
}

// order-preserving f32 <-> u32 encode for atomicMax over signed floats
__device__ __forceinline__ unsigned enc_f32(float f) {
  unsigned u = __float_as_uint(f);
  return (u & 0x80000000u) ? ~u : (u | 0x80000000u);
}
__device__ __forceinline__ float dec_f32(unsigned u) {
  unsigned v = (u & 0x80000000u) ? (u & 0x7fffffffu) : ~u;
  return __uint_as_float(v);
}

// ---- Tensor Data Mover: 1KB contiguous global -> LDS (D# per ISA 08 §8) ----
// data_size=3 (8B units), 1-D tile: tensor_dim0 = tile_dim0 = stride = 128 units.
__device__ __forceinline__ void tdm_load_1kb(unsigned lds_off, const void* gptr) {
  const unsigned long long ga = (unsigned long long)(size_t)gptr;
  const unsigned lo  = (unsigned)__builtin_amdgcn_readfirstlane((int)(unsigned)(ga & 0xffffffffull));
  const unsigned hi  = (unsigned)__builtin_amdgcn_readfirstlane((int)(unsigned)(ga >> 32));
  const unsigned lds = (unsigned)__builtin_amdgcn_readfirstlane((int)lds_off);
  u32x4 g0;
  g0[0] = 1u;                                   // count=1, user-mode descriptor
  g0[1] = lds;                                  // lds_addr (bytes)
  g0[2] = lo;                                   // global_addr[31:0]
  g0[3] = (hi & 0x01ffffffu) | 0x80000000u;     // global_addr[56:32] | type=2 @ [127:126]
  i32x8 g1;
  g1[0] = 0x00030000;   // workgroup_mask=0, data_size=3 (8B)
  g1[1] = 0x00800000;   // tensor_dim0[15:0]=128 @ bits[63:48]
  g1[2] = 0x00010000;   // tensor_dim0[31:16]=0, tensor_dim1[15:0]=1
  g1[3] = 0x00800000;   // tensor_dim1[31:16]=0, tile_dim0=128
  g1[4] = 1;            // tile_dim1=1, tile_dim2=0
  g1[5] = 128;          // tensor_dim0_stride[31:0]=128
  g1[6] = 0x00800000;   // stride0[47:32]=0, tensor_dim1_stride[15:0]=128
  g1[7] = 0;            // tensor_dim1_stride[47:16]=0
  const i32x4 gz = (i32x4){0, 0, 0, 0};
#if defined(__clang_major__) && (__clang_major__ >= 23)
  const i32x8 gz8 = (i32x8){0, 0, 0, 0, 0, 0, 0, 0};
  __builtin_amdgcn_tensor_load_to_lds(g0, g1, gz, gz, gz8, 0);
#else
  __builtin_amdgcn_tensor_load_to_lds(g0, g1, gz, gz, 0);
#endif
}

// ---------------- k0: init stats + encoded max buffer ----------------
__global__ void init_kernel(unsigned* __restrict__ maxbuf, float* __restrict__ stats) {
  int i = blockIdx.x * blockDim.x + threadIdx.x;
  if (i < BB * NN * CO) maxbuf[i] = 0x007FFFFFu;   // enc(-inf)
  if (i < 256) stats[i] = 0.0f;                    // stats1[128] ++ stats2[128]
}

// ---------------- k1: KNN top-20 (incl. self), pd = -||q-p||^2 ----------------
__global__ void knn_kernel(const float* __restrict__ xyz, int* __restrict__ idxout) {
  const int b = blockIdx.y;
  const int n = blockIdx.x * blockDim.x + threadIdx.x;
  const float* X = xyz + (size_t)b * 3 * NN;
  const float qx = X[n], qy = X[NN + n], qz = X[2 * NN + n];
  const float qxx = qx * qx + qy * qy + qz * qz;

  __shared__ float sx[256], sy[256], sz[256], sxx[256];

  float  v[KNB + 1];
  int    id[KNB + 1];
#pragma unroll
  for (int s = 0; s < KNB + 1; ++s) { v[s] = -3.4e38f; id[s] = 0x7fffffff; }

  for (int t = 0; t < NN; t += 256) {
    __syncthreads();
    const int j = t + threadIdx.x;
    const float px = X[j], py = X[NN + j], pz = X[2 * NN + j];
    sx[threadIdx.x] = px; sy[threadIdx.x] = py; sz[threadIdx.x] = pz;
    sxx[threadIdx.x] = px * px + py * py + pz * pz;
    __syncthreads();
    if (t + 256 < NN) {   // gfx1250 global_prefetch_b8 for the next tile
      __builtin_prefetch(&X[t + 256 + threadIdx.x], 0, 1);
      __builtin_prefetch(&X[NN + t + 256 + threadIdx.x], 0, 1);
      __builtin_prefetch(&X[2 * NN + t + 256 + threadIdx.x], 0, 1);
    }
    for (int jj = 0; jj < 256; ++jj) {
      const float d = 2.0f * (qx * sx[jj] + qy * sy[jj] + qz * sz[jj]) - qxx - sxx[jj];
      const int gj = t + jj;
      if (d > v[KNB] || (d == v[KNB] && gj < id[KNB])) {
        float cv = d; int ci = gj;
#pragma unroll
        for (int s = 0; s < KNB + 1; ++s) {
          const bool bet = (cv > v[s]) || (cv == v[s] && ci < id[s]);
          if (bet) { float tv = v[s]; int ti = id[s]; v[s] = cv; id[s] = ci; cv = tv; ci = ti; }
        }
      }
    }
  }
  int* op = idxout + ((size_t)b * NN + n) * KNB;
#pragma unroll
  for (int k = 0; k < KNB; ++k) op[k] = id[k];
}

// ---------------- k2: edge features -> f16, K-dim padded 9 -> 32 ----------------
__global__ void edge_kernel(const float* __restrict__ xyz, const int* __restrict__ idx,
                            _Float16* __restrict__ edge) {
  const long long r = (long long)blockIdx.x * blockDim.x + threadIdx.x;
  if (r >= RTOT) return;
  const long long bn = r / KNB;
  const int n = (int)(bn % NN);
  const int b = (int)(bn / NN);
  const float* X = xyz + (size_t)b * 3 * NN;
  const int j = idx[r];
  const float qx = X[n], qy = X[NN + n], qz = X[2 * NN + n];
  const float px = X[j], py = X[NN + j], pz = X[2 * NN + j];

  alignas(16) _Float16 e[32];
  e[0] = (_Float16)qx; e[1] = (_Float16)qy; e[2] = (_Float16)qz;
  e[3] = (_Float16)px; e[4] = (_Float16)py; e[5] = (_Float16)pz;
  e[6] = (_Float16)(qx - px); e[7] = (_Float16)(qy - py); e[8] = (_Float16)(qz - pz);
#pragma unroll
  for (int c = 9; c < 32; ++c) e[c] = (_Float16)0.0f;

  uint4* dst = (uint4*)(edge + (size_t)r * 32);
  const uint4* src = (const uint4*)e;
  dst[0] = src[0]; dst[1] = src[1]; dst[2] = src[2]; dst[3] = src[3];
}

// ---- A-fragment loader from a 16x32-half row-major tile base (global or LDS) ----
__device__ __forceinline__ v16h load_A_tile(const _Float16* __restrict__ base,
                                            int m, int half) {
  v16h a;
  const _Float16* rowp = base + m * 32;
#pragma unroll
  for (int vv = 0; vv < 8; ++vv) {
    const int K0 = ((vv < 4) ? 2 * vv : 16 + 2 * (vv - 4)) + 8 * half;
    a[2 * vv]     = rowp[K0];
    a[2 * vv + 1] = rowp[K0 + 1];
  }
  return a;
}

// ---------------- k3: GEMM1 (WMMA) stats-only pass ----------------
__global__ void gemm1_stats_kernel(const _Float16* __restrict__ edge,
                                   const float* __restrict__ W1,
                                   float* __restrict__ stats1) {
  const int lane = threadIdx.x & 31;
  const int wave = threadIdx.x >> 5;
  const int gwave = blockIdx.x * (blockDim.x >> 5) + wave;
  const int nwaves = gridDim.x * (blockDim.x >> 5);
  const int m = lane & 15, half = lane >> 4, nloc = lane & 15;

  // B fragments: W1^T per 16-channel tile (K padded 9->32 with zeros)
  v16h bf[4];
#pragma unroll
  for (int ct = 0; ct < 4; ++ct) {
    const int ch = ct * 16 + nloc;
#pragma unroll
    for (int vv = 0; vv < 8; ++vv) {
      const int K0 = ((vv < 4) ? 2 * vv : 16 + 2 * (vv - 4)) + 8 * half;
      bf[ct][2 * vv]     = (K0 < 9)     ? (_Float16)W1[ch * 9 + K0]     : (_Float16)0.0f;
      bf[ct][2 * vv + 1] = (K0 + 1 < 9) ? (_Float16)W1[ch * 9 + K0 + 1] : (_Float16)0.0f;
    }
  }

  float acc_s[4] = {0, 0, 0, 0}, acc_q[4] = {0, 0, 0, 0};
  for (int rt = gwave; rt < RT16; rt += nwaves) {
    const v16h a = load_A_tile(edge + (size_t)rt * 512, m, half);
#pragma unroll
    for (int ct = 0; ct < 4; ++ct) {
      v8f c = {};
      c = wmma_f16(a, bf[ct], c);
#pragma unroll
      for (int vv = 0; vv < 8; ++vv) { const float d = c[vv]; acc_s[ct] += d; acc_q[ct] += d * d; }
    }
  }

  __shared__ float s_s[64], s_q[64];
  if (threadIdx.x < 64) { s_s[threadIdx.x] = 0.0f; s_q[threadIdx.x] = 0.0f; }
  __syncthreads();
#pragma unroll
  for (int ct = 0; ct < 4; ++ct) {
    const int ch = ct * 16 + nloc;
    atomicAdd(&s_s[ch], acc_s[ct]);
    atomicAdd(&s_q[ch], acc_q[ct]);
  }
  __syncthreads();
  if (threadIdx.x < 64) {
    atomicAdd(&stats1[threadIdx.x], s_s[threadIdx.x]);
    atomicAdd(&stats1[64 + threadIdx.x], s_q[threadIdx.x]);
  }
}

// ---------------- finalize BN params: sb[c]=scale, sb[64+c]=bias ----------------
__global__ void finalize_bn_kernel(const float* __restrict__ stats,
                                   const float* __restrict__ g, const float* __restrict__ bvec,
                                   float* __restrict__ sb, float invcnt) {
  const int c = threadIdx.x;
  if (c >= 64) return;
  const float mean = stats[c] * invcnt;
  const float var  = stats[64 + c] * invcnt - mean * mean;
  const float sc   = g[c] * rsqrtf(var + EPS);
  sb[c]      = sc;
  sb[64 + c] = bvec[c] - mean * sc;
}

// ------- k4: fused  GEMM1 -> BN1+LReLU -> GEMM2 (WMMA) -> stats2 + max-over-K -------
// A-tiles are streamed by the Tensor Data Mover into per-wave LDS double buffers,
// overlapped with WMMA on the current tile (TENSORcnt pipelining).
__global__ void fused_mlp_kernel(const _Float16* __restrict__ edge,
                                 const float* __restrict__ W1,
                                 const float* __restrict__ W2,
                                 const float* __restrict__ sb1,
                                 float* __restrict__ stats2,
                                 unsigned* __restrict__ maxbuf) {
  const int lane = threadIdx.x & 31;
  const int wave = threadIdx.x >> 5;
  const int gwave = blockIdx.x * (blockDim.x >> 5) + wave;
  const int nwaves = gridDim.x * (blockDim.x >> 5);
  const int m = lane & 15, half = lane >> 4, nloc = lane & 15;

  __shared__ _Float16 stage[8][2][512];  // per-wave TDM double buffer: 16x32 halves
  __shared__ _Float16 sh[8][16 * 64];    // per-wave 16x64 f16 h1 tile (2 KB each)
  __shared__ float s_s[64], s_q[64];
  _Float16* shw = sh[wave];
  const unsigned stg_off0 = (unsigned)(size_t)&stage[wave][0][0];
  const unsigned stg_off1 = (unsigned)(size_t)&stage[wave][1][0];

  // W1 B-fragments (4 tiles, K padded 9->32)
  v16h bf1[4];
#pragma unroll
  for (int ct = 0; ct < 4; ++ct) {
    const int ch = ct * 16 + nloc;
#pragma unroll
    for (int vv = 0; vv < 8; ++vv) {
      const int K0 = ((vv < 4) ? 2 * vv : 16 + 2 * (vv - 4)) + 8 * half;
      bf1[ct][2 * vv]     = (K0 < 9)     ? (_Float16)W1[ch * 9 + K0]     : (_Float16)0.0f;
      bf1[ct][2 * vv + 1] = (K0 + 1 < 9) ? (_Float16)W1[ch * 9 + K0 + 1] : (_Float16)0.0f;
    }
  }
  // W2 B-fragments (4 col tiles x 2 K-steps of 32)
  v16h bf2[4][2];
#pragma unroll
  for (int ct = 0; ct < 4; ++ct) {
    const int ch = ct * 16 + nloc;
#pragma unroll
    for (int ks = 0; ks < 2; ++ks)
#pragma unroll
      for (int vv = 0; vv < 8; ++vv) {
        const int K0 = ((vv < 4) ? 2 * vv : 16 + 2 * (vv - 4)) + 8 * half + ks * 32;
        bf2[ct][ks][2 * vv]     = (_Float16)W2[ch * 64 + K0];
        bf2[ct][ks][2 * vv + 1] = (_Float16)W2[ch * 64 + K0 + 1];
      }
  }
  // BN1 params per owned channel
  float sc1[4], bi1[4];
#pragma unroll
  for (int ct = 0; ct < 4; ++ct) { sc1[ct] = sb1[ct * 16 + nloc]; bi1[ct] = sb1[64 + ct * 16 + nloc]; }

  float acc_s[4] = {0, 0, 0, 0}, acc_q[4] = {0, 0, 0, 0};

  // TDM prologue: stage first tile
  if (gwave < RT16) tdm_load_1kb(stg_off0, edge + (size_t)gwave * 512);
  int buf = 0;

  for (int rt = gwave; rt < RT16; rt += nwaves) {
    // TDM prefetch of next tile into the other buffer, then wait for current
    const int rtn = rt + nwaves;
    if (rtn < RT16) {
      tdm_load_1kb(buf ? stg_off0 : stg_off1, edge + (size_t)rtn * 512);
      __builtin_amdgcn_s_wait_tensorcnt(1);
    } else {
      __builtin_amdgcn_s_wait_tensorcnt(0);
    }

    // --- GEMM1 from LDS-staged tile ---
    const v16h a = load_A_tile(&stage[wave][buf][0], m, half);
#pragma unroll
    for (int ct = 0; ct < 4; ++ct) {
      v8f z = {};
      const v8f d1 = wmma_f16(a, bf1[ct], z);
      // BN1 + LeakyReLU, scatter D-layout -> row-major LDS tile (in-wave transpose)
#pragma unroll
      for (int vv = 0; vv < 8; ++vv) {
        float h = d1[vv] * sc1[ct] + bi1[ct];
        h = (h >= 0.0f) ? h : SLOPE * h;
        const int row = vv + 8 * half;
        shw[row * 64 + (ct * 16 + nloc)] = (_Float16)h;
      }
    }
    __builtin_amdgcn_wave_barrier();   // DS ops are in-order per wave; fence the scheduler

    // --- GEMM2: K=64 in two 32-steps ---
    v8f d2[4];
#pragma unroll
    for (int ct = 0; ct < 4; ++ct) d2[ct] = (v8f){};
#pragma unroll
    for (int ks = 0; ks < 2; ++ks) {
      v16h a2;
#pragma unroll
      for (int vv = 0; vv < 8; ++vv) {
        const int K0 = ((vv < 4) ? 2 * vv : 16 + 2 * (vv - 4)) + 8 * half + ks * 32;
        a2[2 * vv]     = shw[m * 64 + K0];
        a2[2 * vv + 1] = shw[m * 64 + K0 + 1];
      }
#pragma unroll
      for (int ct = 0; ct < 4; ++ct) d2[ct] = wmma_f16(a2, bf2[ct][ks], d2[ct]);
    }
    __builtin_amdgcn_wave_barrier();

    // --- stats2 + encoded atomic max over K (pre-BN; BN2+LReLU is monotone since g2>0) ---
#pragma unroll
    for (int ct = 0; ct < 4; ++ct) {
      const int ch = ct * 16 + nloc;
      unsigned best = 0; int curp = -1;
#pragma unroll
      for (int vv = 0; vv < 8; ++vv) {
        const float d = d2[ct][vv];
        acc_s[ct] += d; acc_q[ct] += d * d;
        const int gr = rt * 16 + vv + 8 * half;
        const int p  = gr / KNB;
        const unsigned e = enc_f32(d);
        if (p != curp) {
          if (curp >= 0) atomicMax(&maxbuf[(size_t)curp * 64 + ch], best);
          curp = p; best = e;
        } else if (e > best) best = e;
      }
      atomicMax(&maxbuf[(size_t)curp * 64 + ch], best);
    }
    buf ^= 1;
  }

  if (threadIdx.x < 64) { s_s[threadIdx.x] = 0.0f; s_q[threadIdx.x] = 0.0f; }
  __syncthreads();
#pragma unroll
  for (int ct = 0; ct < 4; ++ct) {
    const int ch = ct * 16 + nloc;
    atomicAdd(&s_s[ch], acc_s[ct]);
    atomicAdd(&s_q[ch], acc_q[ct]);
  }
  __syncthreads();
  if (threadIdx.x < 64) {
    atomicAdd(&stats2[threadIdx.x], s_s[threadIdx.x]);
    atomicAdd(&stats2[64 + threadIdx.x], s_q[threadIdx.x]);
  }
}

// ---------------- k5: decode max, apply BN2+LReLU, transpose to [B,Co,N] ----------------
__global__ void output_kernel(const unsigned* __restrict__ maxbuf,
                              const float* __restrict__ sb2,
                              float* __restrict__ out) {
  const long long i = (long long)blockIdx.x * blockDim.x + threadIdx.x;
  if (i >= (long long)BB * CO * NN) return;
  const int n = (int)(i % NN);
  const long long t = i / NN;
  const int ch = (int)(t % CO);
  const int b  = (int)(t / CO);
  const float d = dec_f32(maxbuf[((size_t)b * NN + n) * CO + ch]);
  float h = d * sb2[ch] + sb2[64 + ch];
  out[i] = (h >= 0.0f) ? h : SLOPE * h;
}

// ---------------------------------- host ----------------------------------
extern "C" void kernel_launch(void* const* d_in, const int* in_sizes, int n_in,
                              void* d_out, int out_size, void* d_ws, size_t ws_size,
                              hipStream_t stream) {
  (void)in_sizes; (void)n_in; (void)out_size; (void)ws_size;
  const float* xyz = (const float*)d_in[0];
  const float* W1  = (const float*)d_in[1];
  const float* g1  = (const float*)d_in[2];
  const float* b1  = (const float*)d_in[3];
  const float* W2  = (const float*)d_in[4];
  const float* g2  = (const float*)d_in[5];
  const float* b2  = (const float*)d_in[6];
  float* out = (float*)d_out;

  // workspace layout (all offsets 512B-aligned)
  char* ws = (char*)d_ws;
  int*       idx    = (int*)ws;                                   //  2,621,440 B
  _Float16*  edge   = (_Float16*)(ws + 2621440);                  // 41,943,040 B
  float*     stats  = (float*)(ws + 2621440 + 41943040);          // stats1[128]++stats2[128]
  float*     sb1    = stats + 256;                                // scale1[64]++bias1[64]
  float*     sb2    = sb1 + 128;                                  // scale2[64]++bias2[64]
  unsigned*  maxbuf = (unsigned*)(stats + 512);                   //  8,388,608 B
  float*     stats1 = stats;
  float*     stats2 = stats + 128;

  const float invcnt = 1.0f / (float)RTOT;

  hipLaunchKernelGGL(init_kernel, dim3((BB * NN * CO + 255) / 256), dim3(256), 0, stream,
                     maxbuf, stats);
  hipLaunchKernelGGL(knn_kernel, dim3(NN / 256, BB), dim3(256), 0, stream, xyz, idx);
  hipLaunchKernelGGL(edge_kernel, dim3((RTOT + 255) / 256), dim3(256), 0, stream,
                     xyz, idx, edge);
  hipLaunchKernelGGL(gemm1_stats_kernel, dim3(512), dim3(256), 0, stream,
                     edge, W1, stats1);
  hipLaunchKernelGGL(finalize_bn_kernel, dim3(1), dim3(64), 0, stream,
                     stats1, g1, b1, sb1, invcnt);
  hipLaunchKernelGGL(fused_mlp_kernel, dim3(512), dim3(256), 0, stream,
                     edge, W1, W2, sb1, stats2, maxbuf);
  hipLaunchKernelGGL(finalize_bn_kernel, dim3(1), dim3(64), 0, stream,
                     stats2, g2, b2, sb2, invcnt);
  hipLaunchKernelGGL(output_kernel, dim3((BB * CO * NN + 255) / 256), dim3(256), 0, stream,
                     maxbuf, sb2, out);
}
